// MeshGraphMLPGAT_65077344469239
// MI455X (gfx1250) — compile-verified, matrix operand order checked by reading, not run on hardware
//
#include <hip/hip_runtime.h>
#include <hip/hip_bf16.h>
#include <math.h>

typedef __attribute__((ext_vector_type(16))) _Float16 v16h;
typedef __attribute__((ext_vector_type(8)))  float    v8f;
typedef __attribute__((ext_vector_type(4)))  int      v4i;

#define NSRC  100000
#define NDST  100000
#define NEDGE 1600000
#define NHEAD 4
#define FDIM  256          // NHEAD * 64
#define MPAD  100096       // 128 * 782
#define ORD_NEGINF 0x007FFFFFu

#if __has_builtin(__builtin_amdgcn_global_load_async_to_lds_b128)
#define HAVE_ASYNC_LDS 1
typedef __attribute__((address_space(1))) v4i* gptr_b128;
typedef __attribute__((address_space(3))) v4i* lptr_b128;
#else
#define HAVE_ASYNC_LDS 0
#endif

__device__ __forceinline__ void wait_async_lds() {
#if HAVE_ASYNC_LDS
#if __has_builtin(__builtin_amdgcn_s_wait_asynccnt)
  __builtin_amdgcn_s_wait_asynccnt(0);
#else
  asm volatile("s_wait_asynccnt 0" ::: "memory");
#endif
#endif
}

__device__ __forceinline__ unsigned f2ord(float x) {
  unsigned u = __float_as_uint(x);
  return (u & 0x80000000u) ? ~u : (u | 0x80000000u);
}
__device__ __forceinline__ float ord2f(unsigned k) {
  return (k & 0x80000000u) ? __uint_as_float(k & 0x7FFFFFFFu)
                           : __uint_as_float(~k);
}

// ---------------------------------------------------------------------------
// Tiled WMMA GEMM (f32 inputs):
//   C[M,N] = act( A[M,K] @ W[N,K]^T + bias1 + bias2 )
// block = 256 threads (8 waves), tile 128(M) x 64(N), K stepped by 32.
// f32 in memory, f16 in LDS, f32 accumulate via v_wmma_f32_16x16x32_f16.
// store_f16: write C as _Float16 (feeds next GEMM's async f16 path).
// ---------------------------------------------------------------------------
__launch_bounds__(256)
__global__ void gemm_wmma_f16(const float* __restrict__ A,
                              const float* __restrict__ W,
                              const float* __restrict__ bias1,
                              const float* __restrict__ bias2,
                              void* __restrict__ C,
                              int M_in, int M_store, int N, int K,
                              int act, int store_f16) {
  const int SA = 40, SB = 40;                 // padded LDS strides (halves)
  __shared__ _Float16 As[128 * 40];
  __shared__ _Float16 Bs[64 * 40];

  const int tid  = threadIdx.x;
  const int lane = tid & 31;
  const int wv   = tid >> 5;
  const int m0   = blockIdx.y * 128;
  const int n0   = blockIdx.x * 64;

  const int lrow = lane & 15;
  const int kb   = (lane >> 4) * 8;           // K sub-offset per lane half

  v8f acc[4];
#pragma unroll
  for (int j = 0; j < 4; ++j) acc[j] = v8f{};

  for (int kt = 0; kt < K; kt += 32) {
    // ---- stage A tile: 128 x 32 (each thread: 16 consecutive floats) ----
    {
      const int r  = tid >> 1;
      const int c  = (tid & 1) * 16;
      const int gr = m0 + r;
      float v[16];
      if (gr < M_in) {
        const float4* p = (const float4*)(A + (size_t)gr * K + kt + c);
        float4 f0 = p[0], f1 = p[1], f2 = p[2], f3 = p[3];
        v[0]=f0.x; v[1]=f0.y; v[2]=f0.z;  v[3]=f0.w;
        v[4]=f1.x; v[5]=f1.y; v[6]=f1.z;  v[7]=f1.w;
        v[8]=f2.x; v[9]=f2.y; v[10]=f2.z; v[11]=f2.w;
        v[12]=f3.x;v[13]=f3.y;v[14]=f3.z; v[15]=f3.w;
      } else {
#pragma unroll
        for (int j = 0; j < 16; ++j) v[j] = 0.f;
      }
      _Float16* dst = &As[r * SA + c];
#pragma unroll
      for (int j = 0; j < 16; ++j) dst[j] = (_Float16)v[j];
    }
    // ---- stage B tile: 64 x 32 from W rows n0..n0+63 ----
    {
      const int n = tid >> 2;
      const int c = (tid & 3) * 8;
      const float4* p = (const float4*)(W + (size_t)(n0 + n) * K + kt + c);
      float4 f0 = p[0], f1 = p[1];
      _Float16* dst = &Bs[n * SB + c];
      dst[0]=(_Float16)f0.x; dst[1]=(_Float16)f0.y;
      dst[2]=(_Float16)f0.z; dst[3]=(_Float16)f0.w;
      dst[4]=(_Float16)f1.x; dst[5]=(_Float16)f1.y;
      dst[6]=(_Float16)f1.z; dst[7]=(_Float16)f1.w;
    }
    // prefetch next K-tile while we compute on this one
    if (kt + 32 < K) {
      const int gr = m0 + (tid >> 1);
      if (gr < M_in)
        __builtin_prefetch(A + (size_t)gr * K + kt + 32 + (tid & 1) * 16, 0, 1);
      __builtin_prefetch(W + (size_t)(n0 + (tid >> 2)) * K + kt + 32 + (tid & 3) * 8, 0, 1);
    }
    __syncthreads();

    // ---- A fragment (16x32 f16, ISA 7.12.2 layout) ----
    v16h a;
    {
      const _Float16* p0 = &As[(wv * 16 + lrow) * SA + kb];
      const _Float16* p1 = p0 + 16;
#pragma unroll
      for (int t = 0; t < 8; ++t) { a[t] = p0[t]; a[t + 8] = p1[t]; }
    }
    // ---- 4 B fragments + WMMA ----
#pragma unroll
    for (int j = 0; j < 4; ++j) {
      v16h b;
      const _Float16* q0 = &Bs[(j * 16 + lrow) * SB + kb];
      const _Float16* q1 = q0 + 16;
#pragma unroll
      for (int t = 0; t < 8; ++t) { b[t] = q0[t]; b[t + 8] = q1[t]; }
      acc[j] = __builtin_amdgcn_wmma_f32_16x16x32_f16(
          false, a, false, b, (short)0, acc[j], false, false);
    }
    __syncthreads();
  }

  // ---- epilogue: C layout lane 0-15 -> M 0-7, lane 16-31 -> M 8-15 ----
  const int coff  = lane & 15;
  const int rbase = m0 + wv * 16 + ((lane >> 4) << 3);
#pragma unroll
  for (int j = 0; j < 4; ++j) {
    const int col = n0 + j * 16 + coff;
    float bb = (bias1 ? bias1[col] : 0.f) + (bias2 ? bias2[col] : 0.f);
#pragma unroll
    for (int vg = 0; vg < 8; ++vg) {
      const int row = rbase + vg;
      float val = acc[j][vg] + bb;
      if (act) val = val / (1.f + __expf(-val));          // SiLU
      if (row < M_store) {
        if (store_f16) ((_Float16*)C)[(size_t)row * N + col] = (_Float16)val;
        else           ((float*)C)[(size_t)row * N + col]    = val;
      }
    }
  }
}

// ---------------------------------------------------------------------------
// Tiled WMMA GEMM, f16 inputs (A and W already _Float16, fully padded rows):
//   C[M,N] = A[M,K] @ W[N,K]^T + bias
// Staging is a pure copy -> use async global->LDS (ASYNCcnt) when available.
// No predication in staging: EXEC all-ones for async issue.
// ---------------------------------------------------------------------------
__launch_bounds__(256)
__global__ void gemm_wmma_h16(const _Float16* __restrict__ A,
                              const _Float16* __restrict__ W,
                              const float* __restrict__ bias,
                              float* __restrict__ C,
                              int M_store, int N, int K) {
  const int SA = 40, SB = 40;
  __shared__ _Float16 As[128 * 40];
  __shared__ _Float16 Bs[64 * 40];

  const int tid  = threadIdx.x;
  const int lane = tid & 31;
  const int wv   = tid >> 5;
  const int m0   = blockIdx.y * 128;
  const int n0   = blockIdx.x * 64;
  const int lrow = lane & 15;
  const int kb   = (lane >> 4) * 8;

  // per-thread staging coordinates
  const int ra = tid >> 1, ca = (tid & 1) * 16;   // A: 16 halves (32B)
  const int nb = tid >> 2, cb = (tid & 3) * 8;    // B: 8 halves (16B)

  v8f acc[4];
#pragma unroll
  for (int j = 0; j < 4; ++j) acc[j] = v8f{};

  for (int kt = 0; kt < K; kt += 32) {
    const _Float16* ga = A + (size_t)(m0 + ra) * K + kt + ca;
    const _Float16* gb = W + (size_t)(n0 + nb) * K + kt + cb;
    _Float16* la = &As[ra * SA + ca];
    _Float16* lb = &Bs[nb * SB + cb];
#if HAVE_ASYNC_LDS
    __builtin_amdgcn_global_load_async_to_lds_b128(
        (gptr_b128)ga, (lptr_b128)la, 0, 0);
    __builtin_amdgcn_global_load_async_to_lds_b128(
        (gptr_b128)ga, (lptr_b128)la, 16, 0);
    __builtin_amdgcn_global_load_async_to_lds_b128(
        (gptr_b128)gb, (lptr_b128)lb, 0, 0);
    wait_async_lds();
#else
    uint4 a0 = ((const uint4*)ga)[0];
    uint4 a1 = ((const uint4*)ga)[1];
    uint4 b0 = ((const uint4*)gb)[0];
    ((uint4*)la)[0] = a0;
    ((uint4*)la)[1] = a1;
    ((uint4*)lb)[0] = b0;
#endif
    __syncthreads();

    v16h a;
    {
      const _Float16* p0 = &As[(wv * 16 + lrow) * SA + kb];
      const _Float16* p1 = p0 + 16;
#pragma unroll
      for (int t = 0; t < 8; ++t) { a[t] = p0[t]; a[t + 8] = p1[t]; }
    }
#pragma unroll
    for (int j = 0; j < 4; ++j) {
      v16h b;
      const _Float16* q0 = &Bs[(j * 16 + lrow) * SB + kb];
      const _Float16* q1 = q0 + 16;
#pragma unroll
      for (int t = 0; t < 8; ++t) { b[t] = q0[t]; b[t + 8] = q1[t]; }
      acc[j] = __builtin_amdgcn_wmma_f32_16x16x32_f16(
          false, a, false, b, (short)0, acc[j], false, false);
    }
    __syncthreads();
  }

  const int coff  = lane & 15;
  const int rbase = m0 + wv * 16 + ((lane >> 4) << 3);
#pragma unroll
  for (int j = 0; j < 4; ++j) {
    const int col = n0 + j * 16 + coff;
    float bb = bias ? bias[col] : 0.f;
#pragma unroll
    for (int vg = 0; vg < 8; ++vg) {
      const int row = rbase + vg;
      float val = acc[j][vg] + bb;
      if (row < M_store) C[(size_t)row * N + col] = val;
    }
  }
}

// ---------------------------------------------------------------------------
// Fold proj into fc:  W_sf = fc_W @ proj_W (256x32), b_sf = fc_W @ proj_b,
// and fold attn_r into fc:  w_r[h,k] = sum_d attn_r[h,d] * fc_W[h*64+d, k]
// ---------------------------------------------------------------------------
__global__ void k_fuse(const float* __restrict__ fc_W,
                       const float* __restrict__ proj_W,
                       const float* __restrict__ proj_b,
                       const float* __restrict__ attn_r,
                       float* __restrict__ W_sf, float* __restrict__ b_sf,
                       float* __restrict__ w_r) {
  int idx = blockIdx.x * blockDim.x + threadIdx.x;
  if (idx < 256 * 32) {                      // W_sf
    int j = idx >> 5, i = idx & 31;
    float s = 0.f;
    for (int k = 0; k < 64; ++k) s += fc_W[j * 64 + k] * proj_W[k * 32 + i];
    W_sf[idx] = s;
  } else if (idx < 256 * 32 + 256) {         // b_sf
    int j = idx - 256 * 32;
    float s = 0.f;
    for (int k = 0; k < 64; ++k) s += proj_b[k] * fc_W[j * 64 + k];
    b_sf[j] = s;
  } else if (idx < 256 * 32 + 256 + 256) {   // w_r
    int t = idx - (256 * 32 + 256);
    int h = t >> 6, k = t & 63;
    float s = 0.f;
    for (int d = 0; d < 64; ++d) s += attn_r[h * 64 + d] * fc_W[(h * 64 + d) * 64 + k];
    w_r[t] = s;
  }
}

__global__ void k_cvt_f16(const float* __restrict__ src,
                          _Float16* __restrict__ dst, int n) {
  int idx = blockIdx.x * blockDim.x + threadIdx.x;
  if (idx < n) dst[idx] = (_Float16)src[idx];
}

__global__ void k_init(unsigned* __restrict__ emaxk, float* __restrict__ denom) {
  int idx = blockIdx.x * blockDim.x + threadIdx.x;
  if (idx < NDST * NHEAD) { emaxk[idx] = ORD_NEGINF; denom[idx] = 0.f; }
}

// el[n,h] = sum_d feat_src[n, h*64+d] * attn_l[h,d]
__global__ void k_el(const float* __restrict__ feat_src,
                     const float* __restrict__ attn_l, float* __restrict__ el) {
  int idx = blockIdx.x * blockDim.x + threadIdx.x;
  if (idx >= NSRC * NHEAD) return;
  int n = idx >> 2, h = idx & 3;
  const float4* f = (const float4*)(feat_src + (size_t)n * FDIM + h * 64);
  const float4* a = (const float4*)(attn_l + h * 64);
  float s = 0.f;
#pragma unroll
  for (int t = 0; t < 16; ++t) {
    float4 fv = f[t], av = a[t];
    s += fv.x * av.x + fv.y * av.y + fv.z * av.z + fv.w * av.w;
  }
  el[idx] = s;
}

// er[n,h] = sum_k x_dst[n,k] * w_r[h,k]
__global__ void k_er(const float* __restrict__ x_dst,
                     const float* __restrict__ w_r, float* __restrict__ er) {
  int idx = blockIdx.x * blockDim.x + threadIdx.x;
  if (idx >= NDST * NHEAD) return;
  int n = idx >> 2, h = idx & 3;
  const float4* f = (const float4*)(x_dst + (size_t)n * 64);
  const float4* a = (const float4*)(w_r + h * 64);
  float s = 0.f;
#pragma unroll
  for (int t = 0; t < 16; ++t) {
    float4 fv = f[t], av = a[t];
    s += fv.x * av.x + fv.y * av.y + fv.z * av.z + fv.w * av.w;
  }
  er[idx] = s;
}

// e = leaky_relu(el[src] + er[dst]); segment-max via monotone-uint atomicMax
__global__ void k_logits(const int* __restrict__ esrc, const int* __restrict__ edst,
                         const float* __restrict__ el, const float* __restrict__ er,
                         float* __restrict__ e_buf, unsigned* __restrict__ emaxk) {
  int idx = blockIdx.x * blockDim.x + threadIdx.x;
  if (idx >= NEDGE * NHEAD) return;
  int i = idx >> 2, h = idx & 3;
  int s = esrc[i], d = edst[i];
  float x = el[s * NHEAD + h] + er[d * NHEAD + h];
  float v = (x > 0.f) ? x : 0.2f * x;
  e_buf[idx] = v;
  atomicMax(&emaxk[d * NHEAD + h], f2ord(v));
}

// e_exp = exp(e - max); denom = segment_sum(e_exp)
__global__ void k_exp(const int* __restrict__ edst,
                      const unsigned* __restrict__ emaxk,
                      float* __restrict__ e_buf, float* __restrict__ denom) {
  int idx = blockIdx.x * blockDim.x + threadIdx.x;
  if (idx >= NEDGE * NHEAD) return;
  int i = idx >> 2, h = idx & 3;
  int d = edst[i];
  float m  = ord2f(emaxk[d * NHEAD + h]);
  float ee = __expf(e_buf[idx] - m);
  e_buf[idx] = ee;
  atomicAdd(&denom[d * NHEAD + h], ee);
}

// one wave per edge: 32 lanes x 8 features; rst[d] += alpha * feat_src[s]
__launch_bounds__(256)
__global__ void k_agg(const int* __restrict__ esrc, const int* __restrict__ edst,
                      const float* __restrict__ e_buf, const float* __restrict__ denom,
                      const float* __restrict__ feat_src, float* __restrict__ rst) {
  int edge = (blockIdx.x << 3) + (threadIdx.x >> 5);
  if (edge >= NEDGE) return;
  int lane = threadIdx.x & 31;
  int s = esrc[edge], d = edst[edge];
  int h = lane >> 3;                         // 8 features per lane, same head
  float alpha = e_buf[edge * NHEAD + h] / denom[d * NHEAD + h];
  const float4* fp = (const float4*)(feat_src + (size_t)s * FDIM + lane * 8);
  float4 f0 = fp[0], f1 = fp[1];
  float* out = rst + (size_t)d * FDIM + lane * 8;
  atomicAdd(out + 0, alpha * f0.x); atomicAdd(out + 1, alpha * f0.y);
  atomicAdd(out + 2, alpha * f0.z); atomicAdd(out + 3, alpha * f0.w);
  atomicAdd(out + 4, alpha * f1.x); atomicAdd(out + 5, alpha * f1.y);
  atomicAdd(out + 6, alpha * f1.z); atomicAdd(out + 7, alpha * f1.w);
}

// in-place LayerNorm over 512 cols, one block per row
__launch_bounds__(256)
__global__ void k_ln(float* __restrict__ out, const float* __restrict__ g,
                     const float* __restrict__ b) {
  __shared__ float s1[256], s2[256];
  int row = blockIdx.x, tid = threadIdx.x;
  float* p = out + (size_t)row * 512;
  float x0 = p[tid], x1 = p[tid + 256];
  s1[tid] = x0 + x1;
  s2[tid] = x0 * x0 + x1 * x1;
  __syncthreads();
  for (int s = 128; s > 0; s >>= 1) {
    if (tid < s) { s1[tid] += s1[tid + s]; s2[tid] += s2[tid + s]; }
    __syncthreads();
  }
  float mean = s1[0] * (1.f / 512.f);
  float var  = s2[0] * (1.f / 512.f) - mean * mean;
  float r    = rsqrtf(var + 1e-5f);
  p[tid]       = (x0 - mean) * r * g[tid] + b[tid];
  p[tid + 256] = (x1 - mean) * r * g[tid + 256] + b[tid + 256];
}

// ---------------------------------------------------------------------------
extern "C" void kernel_launch(void* const* d_in, const int* in_sizes, int n_in,
                              void* d_out, int out_size, void* d_ws, size_t ws_size,
                              hipStream_t stream) {
  const float* x_src    = (const float*)d_in[0];
  const float* x_dst    = (const float*)d_in[1];
  const int*   edge_src = (const int*)d_in[2];
  const int*   edge_dst = (const int*)d_in[3];
  const float* proj_W   = (const float*)d_in[4];
  const float* proj_b   = (const float*)d_in[5];
  const float* fc_W     = (const float*)d_in[6];
  const float* attn_l   = (const float*)d_in[7];
  const float* attn_r   = (const float*)d_in[8];
  const float* gat_bias = (const float*)d_in[9];
  const float* res_W    = (const float*)d_in[10];
  const float* res_b    = (const float*)d_in[11];
  const float* W1       = (const float*)d_in[12];
  const float* b1       = (const float*)d_in[13];
  const float* W2       = (const float*)d_in[14];
  const float* b2       = (const float*)d_in[15];
  const float* ln_g     = (const float*)d_in[16];
  const float* ln_b     = (const float*)d_in[17];
  float* out = (float*)d_out;

  // workspace carve-up (float-sized slots)
  float*     ws       = (float*)d_ws;
  float*     feat_src = ws;                                   // MPAD*256 f32
  float*     rst      = feat_src + (size_t)MPAD * FDIM;       // MPAD*256 f32
  _Float16*  h1h      = (_Float16*)(rst + (size_t)MPAD * FDIM);   // MPAD*512 f16
  float*     el       = (float*)(h1h + (size_t)MPAD * 512);   // MPAD*4
  float*     er       = el + (size_t)MPAD * NHEAD;            // MPAD*4
  float*     e_buf    = er + (size_t)MPAD * NHEAD;            // NEDGE*4
  unsigned*  emaxk    = (unsigned*)(e_buf + (size_t)NEDGE * NHEAD); // NDST*4
  float*     denom    = (float*)(emaxk + (size_t)NDST * NHEAD);     // NDST*4
  float*     W_sf     = denom + (size_t)NDST * NHEAD;         // 256*32
  float*     b_sf     = W_sf + 256 * 32;                      // 256
  float*     w_r      = b_sf + 256;                           // 4*64
  _Float16*  w2h      = (_Float16*)(w_r + 256);               // 512*512 f16

  const int TB = 256;
  const int MB = MPAD / 128;                                  // 782

  // 1) fold proj/attn_r into fc weights; pre-convert W2 to f16
  k_fuse<<<(256 * 32 + 512 + TB - 1) / TB, TB, 0, stream>>>(
      fc_W, proj_W, proj_b, attn_r, W_sf, b_sf, w_r);
  k_cvt_f16<<<(512 * 512 + TB - 1) / TB, TB, 0, stream>>>(W2, w2h, 512 * 512);
  // 2) init softmax reduction buffers
  k_init<<<(NDST * NHEAD + TB - 1) / TB, TB, 0, stream>>>(emaxk, denom);
  // 3) feat_src = x_src @ W_sf^T + b_sf       (M=1e5, K=32, N=256)
  gemm_wmma_f16<<<dim3(FDIM / 64, MB), TB, 0, stream>>>(
      x_src, W_sf, b_sf, nullptr, feat_src, NSRC, MPAD, FDIM, 32, 0, 0);
  // 4) rst = x_dst @ res_W^T + res_b + gat_bias   (aggregation base)
  gemm_wmma_f16<<<dim3(FDIM / 64, MB), TB, 0, stream>>>(
      x_dst, res_W, res_b, gat_bias, rst, NDST, MPAD, FDIM, 64, 0, 0);
  // 5) attention logits per node
  k_el<<<(NSRC * NHEAD + TB - 1) / TB, TB, 0, stream>>>(feat_src, attn_l, el);
  k_er<<<(NDST * NHEAD + TB - 1) / TB, TB, 0, stream>>>(x_dst, w_r, er);
  // 6) edge softmax (max, exp/sum), then weighted scatter-add into rst
  k_logits<<<(NEDGE * NHEAD + TB - 1) / TB, TB, 0, stream>>>(
      edge_src, edge_dst, el, er, e_buf, emaxk);
  k_exp<<<(NEDGE * NHEAD + TB - 1) / TB, TB, 0, stream>>>(
      edge_dst, emaxk, e_buf, denom);
  k_agg<<<NEDGE / 8, TB, 0, stream>>>(edge_src, edge_dst, e_buf, denom,
                                      feat_src, rst);
  // 7) MLP1: h1 = silu(rst @ W1^T + b1), stored f16  (K=256, N=512)
  gemm_wmma_f16<<<dim3(512 / 64, MB), TB, 0, stream>>>(
      rst, W1, b1, nullptr, h1h, MPAD, MPAD, 512, FDIM, 1, 1);
  // 8) MLP2: out = h1h @ w2h^T + b2, f16 inputs + async LDS staging
  gemm_wmma_h16<<<dim3(512 / 64, MB), TB, 0, stream>>>(
      h1h, w2h, b2, out, NSRC, 512, 512);
  // 9) LayerNorm in place on d_out
  k_ln<<<NSRC, TB, 0, stream>>>(out, ln_g, ln_b);
}
